// CrossLayerTranscoder_38354057953759
// MI455X (gfx1250) — matrix operand order repaired and checked by reading, not compiled, not run on hardware
//
#include <hip/hip_runtime.h>
#include <hip/hip_bf16.h>

// CrossLayerTranscoder for MI455X (gfx1250, wave32).
// Phase 1: feats[s] = relu(x[s] @ W_enc[s]^T)           [16 GEMMs 2048x512x2048]
// Phase 2: recon[l] = sum_{s<=l} feats[s] @ W_dec[s,l]^T [136 GEMMs 2048x2048x512]
// f32 -> bf16 conversion fused into LDS staging (v_cvt_pk_bf16_f32 when available),
// v_wmma_f32_16x16x32_bf16 with f32 accumulate, f32 output.
// ~653 GFLOP vs ~1.7 GB compulsory traffic -> ~73 us memory roofline at 23.3 TB/s;
// bf16 matrix cores push compute well under that.

typedef __attribute__((ext_vector_type(16))) __bf16 v16bf;
typedef __attribute__((ext_vector_type(8)))  __bf16 v8bf;
typedef __attribute__((ext_vector_type(2)))  __bf16 v2bf;
typedef __attribute__((ext_vector_type(8)))  float  v8f;

#define L_ 16
#define T_ 2048
#define H_ 2048
#define F_ 512

#define BM 128          // workgroup M tile
#define BN 64           // workgroup N tile
#define BK 32           // K step (matches wmma 16x16x32)
#define PADK 40         // padded LDS row stride in bf16 (32 + 8): conflict-free b128 reads
#define NTHREADS 256    // 8 waves

#if defined(__has_builtin)
#if __has_builtin(__builtin_amdgcn_cvt_pk_bf16_f32)
#define HAVE_CVT_PK_BF16 1
#endif
#endif

static __device__ __forceinline__ unsigned pack2(float a, float b) {
#ifdef HAVE_CVT_PK_BF16
  v2bf r = __builtin_amdgcn_cvt_pk_bf16_f32(a, b);     // 1 VALU op, RNE
  return __builtin_bit_cast(unsigned, r);
#else
  unsigned ua = __builtin_bit_cast(unsigned, a);
  unsigned ub = __builtin_bit_cast(unsigned, b);
  ua += 0x7FFFu + ((ua >> 16) & 1u);                   // RNE
  ub += 0x7FFFu + ((ub >> 16) & 1u);
  return (ua >> 16) | (ub & 0xFFFF0000u);
#endif
}

// Stage ROWS x BK f32 tile (global row stride ldg) -> bf16 LDS tile (row stride PADK).
// Coalesced float4 loads, b64 LDS stores.
template<int ROWS>
static __device__ __forceinline__ void stage_tile(const float* __restrict__ gsrc, int ldg,
                                                  __bf16* __restrict__ lds) {
  constexpr int QPR  = BK / 4;               // float4 per row = 8
  constexpr int TOTQ = ROWS * QPR;
  constexpr int ITER = TOTQ / NTHREADS;
  const int tid = threadIdx.x;
#pragma unroll
  for (int i = 0; i < ITER; ++i) {
    const int q   = tid + i * NTHREADS;
    const int row = q / QPR;
    const int kc  = (q % QPR) * 4;
    const float4 v = *reinterpret_cast<const float4*>(gsrc + (size_t)row * ldg + kc);
    uint2 p;
    p.x = pack2(v.x, v.y);
    p.y = pack2(v.z, v.w);
    *reinterpret_cast<uint2*>(lds + row * PADK + kc) = p;
  }
}

// One cacheline prefetch per tile row, two K-steps ahead (rows are BK*4 = 128B).
template<int ROWS>
static __device__ __forceinline__ void prefetch_tile(const float* __restrict__ gsrc, int ldg) {
  const int tid = threadIdx.x;
  if (tid < ROWS)
    __builtin_prefetch(gsrc + (size_t)tid * ldg, 0, 1);
}

// A fragment (16x32 MxK, 16-bit): lane holds row M=lane&15;
// elems 0..7 = K at klo..klo+7, elems 8..15 = K at klo+16..klo+23, klo = (lane<16)?0:8.
static __device__ __forceinline__ v16bf load_frag_a(const __bf16* lds, int mo, int lane) {
  const int row = mo + (lane & 15);
  const int klo = (lane < 16) ? 0 : 8;
  const __bf16* p = lds + row * PADK + klo;
  v8bf lo = *reinterpret_cast<const v8bf*>(p);
  v8bf hi = *reinterpret_cast<const v8bf*>(p + 16);
  return __builtin_shufflevector(lo, hi, 0,1,2,3,4,5,6,7,8,9,10,11,12,13,14,15);
}

// B fragment (32x16 KxN, 16-bit): lane holds column N=lane&15;
// elems 0..7 = K at klo..klo+7, elems 8..15 = K at klo+16..klo+23, klo = (lane<16)?0:8. Source rows are N-major, K-contiguous
// (W_enc[f,h] / W_dec[h,f]) so this is a contiguous read.
static __device__ __forceinline__ v16bf load_frag_b(const __bf16* lds, int no, int lane) {
  const int row  = no + (lane & 15);
  const int koff = (lane < 16) ? 0 : 8;
  const __bf16* p = lds + row * PADK + koff;
  v8bf lo = *reinterpret_cast<const v8bf*>(p);
  v8bf hi = *reinterpret_cast<const v8bf*>(p + 16);
  return __builtin_shufflevector(lo, hi, 0,1,2,3,4,5,6,7,8,9,10,11,12,13,14,15);
}

#define WMMA_BF16(A, B, C) \
  __builtin_amdgcn_wmma_f32_16x16x32_bf16(false, (A), false, (B), (short)0, (C), false, false)

// ---------------- Phase 1: encode + ReLU ----------------
__global__ __launch_bounds__(NTHREADS)
void clt_encode_kernel(const float* __restrict__ x, const float* __restrict__ W_enc,
                       float* __restrict__ feats) {
  __shared__ __bf16 shA[2][BM * PADK];
  __shared__ __bf16 shB[2][BN * PADK];

  const int tm = blockIdx.x;         // T / BM
  const int tn = blockIdx.y;         // F / BN
  const int s  = blockIdx.z;         // layer

  const float* gA = x     + ((size_t)s * T_ + (size_t)tm * BM) * H_;
  const float* gB = W_enc + ((size_t)s * F_ + (size_t)tn * BN) * H_;

  const int lane = threadIdx.x & 31;
  const int wave = threadIdx.x >> 5;
  const int wm = (wave & 3) * 32;
  const int wn = (wave >> 2) * 32;

  v8f acc[2][2] = {};
  constexpr int NK = H_ / BK;        // 64 K-steps

  stage_tile<BM>(gA, H_, shA[0]);
  stage_tile<BN>(gB, H_, shB[0]);
  __syncthreads();

  for (int kk = 0; kk < NK; ++kk) {
    const int cur = kk & 1, nxt = cur ^ 1;
    if (kk + 1 < NK) {
      stage_tile<BM>(gA + (size_t)(kk + 1) * BK, H_, shA[nxt]);
      stage_tile<BN>(gB + (size_t)(kk + 1) * BK, H_, shB[nxt]);
    }
    if (kk + 2 < NK) {
      prefetch_tile<BM>(gA + (size_t)(kk + 2) * BK, H_);
      prefetch_tile<BN>(gB + (size_t)(kk + 2) * BK, H_);
    }
    const v16bf a0 = load_frag_a(shA[cur], wm,      lane);
    const v16bf a1 = load_frag_a(shA[cur], wm + 16, lane);
    const v16bf b0 = load_frag_b(shB[cur], wn,      lane);
    const v16bf b1 = load_frag_b(shB[cur], wn + 16, lane);
    acc[0][0] = WMMA_BF16(a0, b0, acc[0][0]);
    acc[0][1] = WMMA_BF16(a0, b1, acc[0][1]);
    acc[1][0] = WMMA_BF16(a1, b0, acc[1][0]);
    acc[1][1] = WMMA_BF16(a1, b1, acc[1][1]);
    __syncthreads();
  }

  // C layout: VGPR r -> M = r + 8*(lane>=16), N = lane&15. Apply ReLU, store f32.
  float* gC = feats + ((size_t)s * T_ + (size_t)tm * BM) * F_ + (size_t)tn * BN;
  const int m0 = wm + ((lane < 16) ? 0 : 8);
  const int n0 = wn + (lane & 15);
#pragma unroll
  for (int i = 0; i < 2; ++i)
#pragma unroll
    for (int j = 0; j < 2; ++j)
#pragma unroll
      for (int r = 0; r < 8; ++r) {
        float v = acc[i][j][r];
        v = v > 0.0f ? v : 0.0f;
        gC[(size_t)(m0 + i * 16 + r) * F_ + (n0 + j * 16)] = v;
      }
}

// ---------------- Phase 2: causal decode ----------------
__global__ __launch_bounds__(NTHREADS)
void clt_decode_kernel(const float* __restrict__ feats, const float* __restrict__ W_dec,
                       float* __restrict__ recon) {
  __shared__ __bf16 shA[2][BM * PADK];
  __shared__ __bf16 shB[2][BN * PADK];

  const int tm = blockIdx.x;         // T / BM
  const int tn = blockIdx.y;         // H / BN
  const int l  = blockIdx.z;         // target layer

  const int lane = threadIdx.x & 31;
  const int wave = threadIdx.x >> 5;
  const int wm = (wave & 3) * 32;
  const int wn = (wave >> 2) * 32;

  constexpr int KPS = F_ / BK;             // 16 K-steps per source layer
  const int nsteps = (l + 1) * KPS;        // causal: sources s = 0..l

  v8f acc[2][2] = {};

  // step -> (source layer s, k offset inside F)
  auto srcA = [&](int step) {
    const int s  = step >> 4;
    const int k0 = (step & (KPS - 1)) * BK;
    return feats + ((size_t)s * T_ + (size_t)tm * BM) * F_ + k0;
  };
  auto srcB = [&](int step) {
    const int s  = step >> 4;
    const int k0 = (step & (KPS - 1)) * BK;
    return W_dec + (((size_t)s * L_ + l) * H_ + (size_t)tn * BN) * F_ + k0;
  };

  stage_tile<BM>(srcA(0), F_, shA[0]);
  stage_tile<BN>(srcB(0), F_, shB[0]);
  __syncthreads();

  for (int kk = 0; kk < nsteps; ++kk) {
    const int cur = kk & 1, nxt = cur ^ 1;
    if (kk + 1 < nsteps) {
      stage_tile<BM>(srcA(kk + 1), F_, shA[nxt]);
      stage_tile<BN>(srcB(kk + 1), F_, shB[nxt]);
    }
    if (kk + 2 < nsteps) {
      prefetch_tile<BM>(srcA(kk + 2), F_);
      prefetch_tile<BN>(srcB(kk + 2), F_);
    }
    const v16bf a0 = load_frag_a(shA[cur], wm,      lane);
    const v16bf a1 = load_frag_a(shA[cur], wm + 16, lane);
    const v16bf b0 = load_frag_b(shB[cur], wn,      lane);
    const v16bf b1 = load_frag_b(shB[cur], wn + 16, lane);
    acc[0][0] = WMMA_BF16(a0, b0, acc[0][0]);
    acc[0][1] = WMMA_BF16(a0, b1, acc[0][1]);
    acc[1][0] = WMMA_BF16(a1, b0, acc[1][0]);
    acc[1][1] = WMMA_BF16(a1, b1, acc[1][1]);
    __syncthreads();
  }

  float* gC = recon + ((size_t)l * T_ + (size_t)tm * BM) * H_ + (size_t)tn * BN;
  const int m0 = wm + ((lane < 16) ? 0 : 8);
  const int n0 = wn + (lane & 15);
#pragma unroll
  for (int i = 0; i < 2; ++i)
#pragma unroll
    for (int j = 0; j < 2; ++j)
#pragma unroll
      for (int r = 0; r < 8; ++r)
        gC[(size_t)(m0 + i * 16 + r) * H_ + (n0 + j * 16)] = acc[i][j][r];
}

extern "C" void kernel_launch(void* const* d_in, const int* in_sizes, int n_in,
                              void* d_out, int out_size, void* d_ws, size_t ws_size,
                              hipStream_t stream) {
  (void)in_sizes; (void)n_in; (void)out_size; (void)d_ws; (void)ws_size;
  const float* x     = (const float*)d_in[0];   // [L, T, H]
  const float* W_enc = (const float*)d_in[1];   // [L, F, H]
  const float* W_dec = (const float*)d_in[2];   // [L, L, H, F]
  float* feats = (float*)d_out;                       // [L, T, F]
  float* recon = feats + (size_t)L_ * T_ * F_;        // [L, T, H]

  dim3 blk(NTHREADS);
  clt_encode_kernel<<<dim3(T_ / BM, F_ / BN, L_), blk, 0, stream>>>(x, W_enc, feats);
  clt_decode_kernel<<<dim3(T_ / BM, H_ / BN, L_), blk, 0, stream>>>(feats, W_dec, recon);
}
